// DepthAttention_10926396801347
// MI455X (gfx1250) — matrix operand order repaired and checked by reading, not compiled, not run on hardware
//
#include <hip/hip_runtime.h>
#include <hip/hip_bf16.h>

// Problem constants (match reference)
#define Bc 4
#define Hc 8
#define Sc 1024
#define Dc 64
#define KC 256      // k-chunk rows staged in LDS
#define PADK 8      // LDS pad (bf16 elems), keeps 16B row alignment
#define ROWS 128    // q-rows per block (8 waves * 16)
#define NTHREADS 256

typedef __bf16 bf16_t;
typedef __attribute__((ext_vector_type(16))) __bf16 v16bf;
typedef __attribute__((ext_vector_type(8)))  __bf16 bf16x8;
typedef __attribute__((ext_vector_type(8)))  float  v8f;
typedef __attribute__((ext_vector_type(8)))  float  f32x8;

struct Smem {
    union {
        alignas(16) bf16_t kbuf[KC * Dc];            // pass 1: K chunk, row-major
        alignas(16) bf16_t vt[Dc * (KC + PADK)];     // pass 2: V chunk, transposed [d][k]
    } u;
    float mrow[8][16];
    float lrow[8][16];
};

__device__ __forceinline__ float fastrcp(float x) { return __builtin_amdgcn_rcpf(x); }

__device__ __forceinline__ v16bf pack16(bf16x8 lo, bf16x8 hi) {
    return __builtin_shufflevector(lo, hi, 0,1,2,3,4,5,6,7,8,9,10,11,12,13,14,15);
}

__global__ __launch_bounds__(NTHREADS)
void depth_attn_kernel(const float* __restrict__ Qg, const float* __restrict__ Kg,
                       const float* __restrict__ Vg, const unsigned char* __restrict__ Mg,
                       const float* __restrict__ Dg,
                       float* __restrict__ OutV, float* __restrict__ OutA)
{
    __shared__ Smem smem;

    const int tid  = threadIdx.x;
    const int wave = tid >> 5;
    const int lane = tid & 31;
    const int ln   = lane & 15;      // column index within 16-lane group
    const int hh   = lane >> 4;      // which 16-lane half of the wave

    const int qblocks = Sc / ROWS;                 // 8
    const int bh = blockIdx.x / qblocks;           // fused batch*head: 0..31
    const int qb = blockIdx.x % qblocks;
    const int q0 = qb * ROWS + wave * 16;          // this wave's 16-row q-tile

    const size_t bhSD = (size_t)bh * Sc * Dc;
    const size_t bhSS = (size_t)bh * Sc * Sc;
    const float* Q  = Qg + bhSD;
    const float* K  = Kg + bhSD;
    const float* V  = Vg + bhSD;
    const float* DP = Dg + bhSS;
    const unsigned char* MK = Mg + bhSS;
    float* OV = OutV + bhSD;
    float* OA = OutA + bhSS;

    // ---- Q tile -> bf16 A-layout registers (16x64 = two 16x32 operands) ----
    // A layout: lane holds row ln; elems 0..7 <-> K c0+hh*8+0..7, elems 8..15 <-> +16
    const int qrow = q0 + ln;
    const float* qp = Q + (size_t)qrow * Dc;
    f32x8 qf0 = *(const f32x8*)(qp      + hh * 8);
    f32x8 qf1 = *(const f32x8*)(qp + 16 + hh * 8);
    f32x8 qf2 = *(const f32x8*)(qp + 32 + hh * 8);
    f32x8 qf3 = *(const f32x8*)(qp + 48 + hh * 8);
    v16bf qa0 = pack16(__builtin_convertvector(qf0, bf16x8), __builtin_convertvector(qf1, bf16x8));
    v16bf qa1 = pack16(__builtin_convertvector(qf2, bf16x8), __builtin_convertvector(qf3, bf16x8));

    // Per-lane online softmax state over this lane's column subset (no shuffles in hot loop)
    float mi[8], li[8];
    #pragma unroll
    for (int i = 0; i < 8; ++i) { mi[i] = -3.0e38f; li[i] = 0.0f; }

    // ================= Pass 1: scores + per-lane online softmax =================
    for (int c = 0; c < Sc / KC; ++c) {
        const int kbase = c * KC;
        __syncthreads();
        for (int idx = tid; idx < KC * Dc; idx += NTHREADS)
            smem.u.kbuf[idx] = (bf16_t)K[(size_t)kbase * Dc + idx];
        __syncthreads();

        // prefetch next K chunk into cache while we compute on this one
        if (c + 1 < Sc / KC) {
            const float* knext = K + (size_t)(kbase + KC) * Dc;
            __builtin_prefetch(knext + tid * 64, 0, 0);        // -> global_prefetch_b8
            __builtin_prefetch(knext + tid * 64 + 32, 0, 0);
        }

        for (int kt = 0; kt < KC / 16; ++kt) {
            // B = K^T (64x16): lane holds column n=ln -> K-row kt*16+ln; elems 0..7 <-> d = db+hh*8+0..7, elems 8..15 <-> +16
            const bf16_t* kr = &smem.u.kbuf[(kt * 16 + ln) * Dc];
            v16bf b0 = pack16(*(const bf16x8*)(kr +      hh * 8),
                              *(const bf16x8*)(kr + 16 + hh * 8));
            v16bf b1 = pack16(*(const bf16x8*)(kr + 32 + hh * 8),
                              *(const bf16x8*)(kr + 48 + hh * 8));
            v8f cacc = {};
            cacc = __builtin_amdgcn_wmma_f32_16x16x32_bf16(false, qa0, false, b0, (short)0, cacc, false, false);
            cacc = __builtin_amdgcn_wmma_f32_16x16x32_bf16(false, qa1, false, b1, (short)0, cacc, false, false);

            const int col = kbase + kt * 16 + ln;   // C layout: lane%16 = column
            #pragma unroll
            for (int i = 0; i < 8; ++i) {
                const int row = q0 + i + hh * 8;    // C layout: VGPR i, half -> row
                float sv = cacc[i] * 0.125f;        // 1/sqrt(64)
                const float dep = DP[(size_t)row * Sc + col];
                sv *= fastrcp(1.0f + __expf(-dep)); // sigmoid gate (v_rcp_f32)
                if (MK[(size_t)row * Sc + col]) sv = -3.0e38f;
                OA[(size_t)row * Sc + col] = sv;    // raw gated/masked score (L2-hot)
                // per-lane online update
                const float mnew = fmaxf(mi[i], sv);
                li[i] = li[i] * __expf(mi[i] - mnew) + __expf(sv - mnew);
                mi[i] = mnew;
            }
        }
    }

    // ---- merge (m,l) across the 16 lanes that hold each row (once, not per tile) ----
    #pragma unroll
    for (int i = 0; i < 8; ++i) {
        float m = mi[i], l = li[i];
        #pragma unroll
        for (int off = 1; off < 16; off <<= 1) {
            const float mo = __shfl_xor(m, off, 16);
            const float lo = __shfl_xor(l, off, 16);
            const float mn = fmaxf(m, mo);
            l = l * __expf(m - mn) + lo * __expf(mo - mn);
            m = mn;
        }
        mi[i] = m; li[i] = l;
    }

    // Exchange (m,l): pass 2 A-layout needs per-lane row = ln
    if (ln == 0) {
        #pragma unroll
        for (int i = 0; i < 8; ++i) {
            smem.mrow[wave][hh * 8 + i] = mi[i];
            smem.lrow[wave][hh * 8 + i] = li[i];
        }
    }
    __threadfence_block();   // make pass-1 global score stores visible to pass-2 loads
    __syncthreads();
    const float mr   = smem.mrow[wave][ln];
    const float linv = fastrcp(smem.lrow[wave][ln]);

    // ================= Pass 2: normalize p_attn + P@V via WMMA =================
    v8f acc0 = {}, acc1 = {}, acc2 = {}, acc3 = {};
    float* srow = OA + (size_t)qrow * Sc;   // this lane's score row (A-layout: lane = row)

    for (int c = 0; c < Sc / KC; ++c) {
        const int kbase = c * KC;
        __syncthreads();
        // stage V chunk transposed: vt[d][k]
        for (int idx = tid; idx < KC * Dc; idx += NTHREADS) {
            const int kr2 = idx >> 6, d = idx & 63;
            smem.u.vt[d * (KC + PADK) + kr2] = (bf16_t)V[(size_t)(kbase + kr2) * Dc + d];
        }
        __syncthreads();

        if (c + 1 < Sc / KC) {
            const float* vnext = V + (size_t)(kbase + KC) * Dc;
            __builtin_prefetch(vnext + tid * 64, 0, 0);
            __builtin_prefetch(vnext + tid * 64 + 32, 0, 0);
        }

        for (int kt = 0; kt < KC / 32; ++kt) {
            const int kb = kbase + kt * 32;
            // read raw scores in A-layout chunks, normalize, write final p_attn back
            f32x8 s0 = *(const f32x8*)(srow + kb      + hh * 8);
            f32x8 s1 = *(const f32x8*)(srow + kb + 16 + hh * 8);
            f32x8 p0, p1;
            #pragma unroll
            for (int j = 0; j < 8; ++j) {
                p0[j] = __expf(s0[j] - mr) * linv;
                p1[j] = __expf(s1[j] - mr) * linv;
            }
            *(f32x8*)(srow + kb      + hh * 8) = p0;
            *(f32x8*)(srow + kb + 16 + hh * 8) = p1;
            v16bf pa = pack16(__builtin_convertvector(p0, bf16x8),
                              __builtin_convertvector(p1, bf16x8));

            // B = V (32x16 per d-tile): lane col = dt*16+ln -> vt row; elems 0..7 <-> k = kb+hh*8+0..7, elems 8..15 <-> +16
            const int vo = kt * 32 + hh * 8;
            const bf16_t* v0 = &smem.u.vt[( 0 + ln) * (KC + PADK) + vo];
            const bf16_t* v1 = &smem.u.vt[(16 + ln) * (KC + PADK) + vo];
            const bf16_t* v2 = &smem.u.vt[(32 + ln) * (KC + PADK) + vo];
            const bf16_t* v3 = &smem.u.vt[(48 + ln) * (KC + PADK) + vo];
            v16bf vb0 = pack16(*(const bf16x8*)v0, *(const bf16x8*)(v0 + 16));
            v16bf vb1 = pack16(*(const bf16x8*)v1, *(const bf16x8*)(v1 + 16));
            v16bf vb2 = pack16(*(const bf16x8*)v2, *(const bf16x8*)(v2 + 16));
            v16bf vb3 = pack16(*(const bf16x8*)v3, *(const bf16x8*)(v3 + 16));
            acc0 = __builtin_amdgcn_wmma_f32_16x16x32_bf16(false, pa, false, vb0, (short)0, acc0, false, false);
            acc1 = __builtin_amdgcn_wmma_f32_16x16x32_bf16(false, pa, false, vb1, (short)0, acc1, false, false);
            acc2 = __builtin_amdgcn_wmma_f32_16x16x32_bf16(false, pa, false, vb2, (short)0, acc2, false, false);
            acc3 = __builtin_amdgcn_wmma_f32_16x16x32_bf16(false, pa, false, vb3, (short)0, acc3, false, false);
        }
    }

    // ---- store p_val (C layout: VGPR i -> row, lane%16 -> col within d-tile) ----
    #pragma unroll
    for (int i = 0; i < 8; ++i) {
        const int row = q0 + i + hh * 8;
        float* op = OV + (size_t)row * Dc + ln;
        op[0]  = acc0[i];
        op[16] = acc1[i];
        op[32] = acc2[i];
        op[48] = acc3[i];
    }
}

extern "C" void kernel_launch(void* const* d_in, const int* in_sizes, int n_in,
                              void* d_out, int out_size, void* d_ws, size_t ws_size,
                              hipStream_t stream) {
    (void)in_sizes; (void)n_in; (void)d_ws; (void)ws_size; (void)out_size;
    const float* query = (const float*)d_in[0];
    const float* key   = (const float*)d_in[1];
    const float* value = (const float*)d_in[2];
    const unsigned char* mask = (const unsigned char*)d_in[3];  // jax bool -> 1 byte
    const float* depth = (const float*)d_in[4];

    float* out_val  = (float*)d_out;                                  // [B,H,S,D]
    float* out_attn = (float*)d_out + (size_t)Bc * Hc * Sc * Dc;      // [B,H,S,S]

    const int grid = Bc * Hc * (Sc / ROWS);   // 256 blocks, 8 waves each
    depth_attn_kernel<<<grid, NTHREADS, 0, stream>>>(query, key, value, mask, depth,
                                                     out_val, out_attn);
}
